// MultiHeadEdgeAttention_13469017440326
// MI455X (gfx1250) — compile-verified, matrix operand order checked by reading, not run on hardware
//
#include <hip/hip_runtime.h>
#include <hip/hip_bf16.h>
#include <math.h>

#define B_  4
#define L_  512
#define D_  768
#define H_  12
#define DK_ 64
#define DE_ 64

typedef __attribute__((ext_vector_type(16))) __bf16 v16bf;
typedef __attribute__((ext_vector_type(2)))  __bf16 v2bf;
typedef __attribute__((ext_vector_type(8)))  float  v8f;

union BF16x16 { v16bf v; unsigned int u[8]; };

#if defined(__has_builtin)
#if __has_builtin(__builtin_amdgcn_cvt_pk_bf16_f32)
#define HAVE_CVT_PK_BF16 1
#endif
#endif

// native RTNE f32->bf16
__device__ __forceinline__ unsigned short f2bf(float f) {
  union { __bf16 h; unsigned short s; } v;
  v.h = (__bf16)f;
  return v.s;
}
// pack two f32 -> bf16x2 dword (single v_cvt_pk_bf16_f32 when available)
__device__ __forceinline__ unsigned int pkf(float a, float b) {
#ifdef HAVE_CVT_PK_BF16
  union { v2bf h; unsigned int u; } v;
  v.h = __builtin_amdgcn_cvt_pk_bf16_f32(a, b);
  return v.u;
#else
  union { __bf16 h[2]; unsigned int u; } v;
  v.h[0] = (__bf16)a; v.h[1] = (__bf16)b;
  return v.u;
#endif
}
__device__ __forceinline__ unsigned int pk2(const float* p) { return pkf(p[0], p[1]); }

// ISA 16-bit A/B fragment K pattern: dword j of lane (kh = lane>>4) holds K = kpat, kpat+1
__device__ __forceinline__ int kpat(int j, int kh) {
  return 16 * (j >> 2) + 8 * kh + 2 * (j & 3);
}
__device__ __forceinline__ v8f wmma_bf16(const BF16x16& a, const BF16x16& b, v8f c) {
  return __builtin_amdgcn_wmma_f32_16x16x32_bf16(false, a.v, false, b.v, (short)0, c,
                                                 false, false);
}

// CDNA5 async global->LDS copy (ASYNCcnt-tracked, no VGPR round trip).
// GVS mode: mem = saddr(SGPR pair) + voff(i32); LDS dest = per-lane lds byte addr.
__device__ __forceinline__ void async_ld_b128(unsigned int lds_addr, unsigned int voff,
                                              const void* base) {
  asm volatile("global_load_async_to_lds_b128 %0, %1, %2"
               :: "v"(lds_addr), "v"(voff), "s"(base) : "memory");
}
__device__ __forceinline__ void wait_asynccnt0() {
  asm volatile("s_wait_asynccnt 0" ::: "memory");
}
// low 32 bits of a flat LDS pointer = wave-relative LDS byte address
__device__ __forceinline__ unsigned int lds_addr_of(const void* p) {
  return (unsigned int)(unsigned long long)p;
}

// ---------------------------------------------------------------------------
// Generic GEMM: out = X[M,K] @ W[K,N] + bias, bf16 WMMA, fp32 accum.
// mode 0: store bf16 at (B,H,L,DK)   (q/k head split)
// mode 1: store bf16 at (B,H,DK,L)   (v transposed for P@V B-fragments)
// mode 2: store fp32 at out[row*ld_out + col_off + col]
// Block: 256 thr (8 waves), tile 64(M) x 128(N), K step 32.
// ---------------------------------------------------------------------------
__global__ __launch_bounds__(256)
void gemm_bias_kernel(const float* __restrict__ X, int ldx,
                      const float* __restrict__ W, int N, int K,
                      const float* __restrict__ bias,
                      int mode, void* __restrict__ outp,
                      int ld_out, int col_off) {
  __shared__ unsigned short Xs[64][32];    // rows x k (bf16)
  __shared__ unsigned short Wt[128][32];   // cols x k (bf16, transposed)
  const int tid  = threadIdx.x;
  const int lane = tid & 31;
  const int w    = tid >> 5;
  const int kh   = lane >> 4;
  const int lm   = lane & 15;
  const int r0   = blockIdx.y * 64;
  const int c0   = blockIdx.x * 128;

  v8f acc[4];
  for (int i = 0; i < 4; ++i)
    for (int e = 0; e < 8; ++e) acc[i][e] = 0.0f;

  const int nK = K >> 5;
  for (int kc = 0; kc < nK; ++kc) {
    const int k0 = kc << 5;
    if (kc + 1 < nK) {           // prefetch next K chunk
      __builtin_prefetch(&X[(size_t)(r0 + (tid & 63)) * ldx + k0 + 32], 0, 3);
      __builtin_prefetch(&W[(size_t)(k0 + 32 + (tid & 31)) * N + c0 + ((tid >> 5) << 4)], 0, 3);
    }
#pragma unroll
    for (int i = 0; i < 4; ++i) {               // 64x32: 1024 bf16x2 pairs
      int p = tid + (i << 8);
      int mr = p >> 4, kk = (p & 15) << 1;
      const float* xp = &X[(size_t)(r0 + mr) * ldx + k0 + kk];
      *(unsigned int*)&Xs[mr][kk] = pkf(xp[0], xp[1]);
    }
#pragma unroll
    for (int i = 0; i < 8; ++i) {               // 32x128: 2048 bf16x2 pairs
      int p = tid + (i << 8);
      int nn = p & 127, kk = (p >> 7) << 1;
      *(unsigned int*)&Wt[nn][kk] = pkf(W[(size_t)(k0 + kk) * N + c0 + nn],
                                        W[(size_t)(k0 + kk + 1) * N + c0 + nn]);
    }
    __syncthreads();

    BF16x16 A;
    const int arow = ((w & 3) << 4) + lm;
#pragma unroll
    for (int j = 0; j < 8; ++j)
      A.u[j] = *(const unsigned int*)&Xs[arow][kpat(j, kh)];
    const int cbase = (w >> 2) << 6;
#pragma unroll
    for (int nt = 0; nt < 4; ++nt) {
      BF16x16 Bf;
      const int bn = cbase + (nt << 4) + lm;
#pragma unroll
      for (int j = 0; j < 8; ++j)
        Bf.u[j] = *(const unsigned int*)&Wt[bn][kpat(j, kh)];
      acc[nt] = wmma_bf16(A, Bf, acc[nt]);
    }
    __syncthreads();
  }

#pragma unroll
  for (int nt = 0; nt < 4; ++nt) {
    const int col = c0 + ((w >> 2) << 6) + (nt << 4) + lm;
    const float bv = bias[col];
    const int rbase = r0 + ((w & 3) << 4) + (kh << 3);
    if (mode == 1) {            // rows contiguous in (B,H,DK,L): one b128/lane
      int h = col >> 6, d = col & 63, b = rbase >> 9, nq = rbase & 511;
      uint4 o;
      o.x = pkf(acc[nt][0] + bv, acc[nt][1] + bv);
      o.y = pkf(acc[nt][2] + bv, acc[nt][3] + bv);
      o.z = pkf(acc[nt][4] + bv, acc[nt][5] + bv);
      o.w = pkf(acc[nt][6] + bv, acc[nt][7] + bv);
      *(uint4*)&((unsigned short*)outp)[((((size_t)b * H_ + h) * DK_ + d) << 9) + nq] = o;
    } else {
#pragma unroll
      for (int r = 0; r < 8; ++r) {
        const int row = rbase + r;
        const float val = acc[nt][r] + bv;
        if (mode == 0) {
          int h = col >> 6, d = col & 63, b = row >> 9, nq = row & 511;
          ((unsigned short*)outp)[((((size_t)b * H_ + h) * L_ + nq) << 6) + d] = f2bf(val);
        } else {
          ((float*)outp)[(size_t)row * ld_out + col_off + col] = val;
        }
      }
    }
  }
}

// ---------------------------------------------------------------------------
// Edge kernel: k_edge = edge @ Wke + bke (bf16, stored transposed
// (b,n,DE,Lm)), and eb = 5*tanh((edge@We+be)*2^-0.5 / 5) fp32 (b,n,m).
// Block = 256 thr (8 waves), 64 flat rows (fixed b,n; m0..m0+63).
// 16 KB fp32 edge tile staged with GLOBAL_LOAD_ASYNC_TO_LDS_B128 (ASYNCcnt),
// converted to bf16 only at fragment build; eb reads the fp32 LDS tile.
// ---------------------------------------------------------------------------
__global__ __launch_bounds__(256)
void edge_kernel(const float* __restrict__ E, const float* __restrict__ Wke,
                 const float* __restrict__ bke, const float* __restrict__ Wev,
                 const float* __restrict__ bev,
                 unsigned short* __restrict__ ket, float* __restrict__ eb) {
  __shared__ float          Ef[64][64];    // local m x d (fp32, async-filled)
  __shared__ unsigned short Wkt[64][64];   // out col x d (bf16, transposed)
  const int tid  = threadIdx.x;
  const int lane = tid & 31;
  const int w    = tid >> 5;
  const int kh   = lane >> 4;
  const int lm   = lane & 15;
  const long flat = (long)blockIdx.x * 64;
  const int b  = (int)(flat >> 18);
  const int n  = (int)((flat >> 9) & 511);
  const int m0 = (int)(flat & 511);

  // async global->LDS: 64x64 fp32 = 16 KB = 1024 x b128; 4 per thread
  {
    const unsigned int gbase = (unsigned int)(flat * 256);   // byte offset of tile
    const unsigned int lbase = lds_addr_of(&Ef[0][0]);
#pragma unroll
    for (int i = 0; i < 4; ++i) {
      const unsigned int f = (unsigned int)tid + (i << 8);   // 16-byte chunk id
      async_ld_b128(lbase + f * 16u, gbase + f * 16u, (const void*)E);
    }
  }
#pragma unroll
  for (int i = 0; i < 8; ++i) {            // Wke -> bf16 transposed (paired)
    int p = tid + (i << 8);
    int nn = p & 63, kk = (p >> 6) << 1;
    *(unsigned int*)&Wkt[nn][kk] = pkf(Wke[kk * 64 + nn], Wke[(kk + 1) * 64 + nn]);
  }
  wait_asynccnt0();
  __syncthreads();

  const int mtile = w >> 1;
  BF16x16 A0, A1;
  const int arow = (mtile << 4) + lm;
#pragma unroll
  for (int j = 0; j < 8; ++j) {            // cvt_pk fp32 LDS -> bf16 fragments
    A0.u[j] = pk2(&Ef[arow][kpat(j, kh)]);
    A1.u[j] = pk2(&Ef[arow][32 + kpat(j, kh)]);
  }
#pragma unroll
  for (int t = 0; t < 2; ++t) {
    const int ntile = ((w & 1) << 1) + t;
    BF16x16 B0, B1;
    const int bn = (ntile << 4) + lm;
#pragma unroll
    for (int j = 0; j < 8; ++j) {
      B0.u[j] = *(const unsigned int*)&Wkt[bn][kpat(j, kh)];
      B1.u[j] = *(const unsigned int*)&Wkt[bn][32 + kpat(j, kh)];
    }
    v8f c;
    for (int e = 0; e < 8; ++e) c[e] = 0.0f;
    c = wmma_bf16(A0, B0, c);
    c = wmma_bf16(A1, B1, c);
    const int col = (ntile << 4) + lm;
    const float bv = bke[col];
    uint4 o;                                // 8 contiguous m -> one b128 store
    o.x = pkf(c[0] + bv, c[1] + bv);
    o.y = pkf(c[2] + bv, c[3] + bv);
    o.z = pkf(c[4] + bv, c[5] + bv);
    o.w = pkf(c[6] + bv, c[7] + bv);
    *(uint4*)&ket[((((size_t)b * 512 + n) * 64 + col) << 9) +
                  m0 + (mtile << 4) + (kh << 3)] = o;
  }

  if (tid < 64) {   // edge-bias scalar from the fp32 LDS tile
    float s = bev[0];
    for (int d = 0; d < 64; ++d) s += Ef[tid][d] * Wev[d];
    s *= 0.70710678118654752f;
    s = 5.0f * tanhf(s * 0.2f);
    eb[((size_t)b * 512 + n) * 512 + m0 + tid] = s;
  }
}

// ---------------------------------------------------------------------------
// Attention: 1 wave handles 16 query rows of one (b,h).
// eb (32 KB) and mask (8 KB) for the 16 rows are contiguous -> async DMA to
// LDS; scores accumulate on top of eb in LDS. Masked fp32 softmax, then
// ctx = P @ V (WMMA) -> concat[:, :768]; P (bf16) -> Pws[(b,n),h,m].
// ---------------------------------------------------------------------------
__global__ __launch_bounds__(32)
void attn_kernel(const unsigned short* __restrict__ qw,
                 const unsigned short* __restrict__ kw,
                 const unsigned short* __restrict__ vw,
                 const float* __restrict__ eb,
                 const unsigned char* __restrict__ mask,
                 unsigned short* __restrict__ Pws,
                 float* __restrict__ concat) {
  __shared__ float          S[16][512];
  __shared__ unsigned char  Mk[16][512];
  __shared__ unsigned short Pl[16][512];
  __shared__ float          redM[2][16];
  __shared__ float          redS[2][16];
  const int lane = threadIdx.x & 31;
  const int kh   = lane >> 4;
  const int lm   = lane & 15;
  const int bid  = blockIdx.x;        // B*H*(L/16)
  const int nblk = bid & 31;
  const int bh   = bid >> 5;
  const int h    = bh % H_;
  const int b    = bh / H_;
  const int n0   = nblk << 4;

  // async-stage eb rows (contiguous 32 KB) into S, mask rows (8 KB) into Mk
  {
    const float* ebp = eb + ((size_t)b * 512 + n0) * 512;
    const unsigned char* mkp = mask + ((size_t)b * 512 + n0) * 512;
    const unsigned int sb = lds_addr_of(&S[0][0]);
    const unsigned int mb = lds_addr_of(&Mk[0][0]);
    for (int i = 0; i < 64; ++i) {
      unsigned int f = (unsigned int)lane + (i << 5);
      async_ld_b128(sb + f * 16u, f * 16u, ebp);
    }
    for (int i = 0; i < 16; ++i) {
      unsigned int f = (unsigned int)lane + (i << 5);
      async_ld_b128(mb + f * 16u, f * 16u, mkp);
    }
  }

  BF16x16 A0, A1;   // q fragments (overlaps with async DMA)
  {
    const unsigned short* qrow = qw + ((size_t)bh * 512 + n0 + lm) * 64;
#pragma unroll
    for (int j = 0; j < 8; ++j) {
      A0.u[j] = *(const unsigned int*)&qrow[kpat(j, kh)];
      A1.u[j] = *(const unsigned int*)&qrow[32 + kpat(j, kh)];
    }
  }
  wait_asynccnt0();
  __syncthreads();

  const float scale = 0.08838834764831845f;   // (2*DK)^-0.5
  for (int mt = 0; mt < 32; ++mt) {
    const int m0 = mt << 4;
    BF16x16 B0, B1;
    const unsigned short* krow = kw + ((size_t)bh * 512 + m0 + lm) * 64;
#pragma unroll
    for (int j = 0; j < 8; ++j) {
      B0.u[j] = *(const unsigned int*)&krow[kpat(j, kh)];
      B1.u[j] = *(const unsigned int*)&krow[32 + kpat(j, kh)];
    }
    v8f c;
    for (int e = 0; e < 8; ++e) c[e] = 0.0f;
    c = wmma_bf16(A0, B0, c);
    c = wmma_bf16(A1, B1, c);
#pragma unroll
    for (int r = 0; r < 8; ++r) {
      const int M  = r + (kh << 3);
      const int mk = m0 + lm;
      float v = c[r] * scale + S[M][mk];
      if (Mk[M][mk]) v = -1e9f;
      S[M][mk] = v;
    }
  }
  __syncthreads();

  // row-wise softmax: 2 lanes per row (256 cols each)
  const int row = lm;
  const int cb  = kh << 8;
  float mx = -3.0e38f;
  for (int ci = 0; ci < 256; ++ci) mx = fmaxf(mx, S[row][cb + ci]);
  redM[kh][row] = mx;
  __syncthreads();
  const float gm = fmaxf(redM[0][row], redM[1][row]);
  float sum = 0.0f;
  for (int ci = 0; ci < 256; ++ci) {
    float e = __expf(S[row][cb + ci] - gm);
    S[row][cb + ci] = e;
    sum += e;
  }
  redS[kh][row] = sum;
  __syncthreads();
  const float inv = 1.0f / (redS[0][row] + redS[1][row]);
  unsigned short* pg = Pws + (((size_t)b * 512 + n0 + row) * 16 + h) * 512 + cb;
  for (int ci = 0; ci < 256; ci += 2) {
    unsigned int pk = pkf(S[row][cb + ci] * inv, S[row][cb + ci + 1] * inv);
    *(unsigned int*)&Pl[row][cb + ci] = pk;
    *(unsigned int*)&pg[ci] = pk;
  }
  __syncthreads();

  // ctx = P @ V  (V stored transposed (B,H,DK,L))
  v8f cc[4];
  for (int i = 0; i < 4; ++i)
    for (int e = 0; e < 8; ++e) cc[i][e] = 0.0f;
  for (int kc = 0; kc < 16; ++kc) {
    const int kk0 = kc << 5;
    BF16x16 A;
#pragma unroll
    for (int j = 0; j < 8; ++j)
      A.u[j] = *(const unsigned int*)&Pl[lm][kk0 + kpat(j, kh)];
#pragma unroll
    for (int nt = 0; nt < 4; ++nt) {
      const int d = (nt << 4) + lm;
      const unsigned short* vrow = vw + ((size_t)bh * 64 + d) * 512;
      BF16x16 Bf;
#pragma unroll
      for (int j = 0; j < 8; ++j)
        Bf.u[j] = *(const unsigned int*)&vrow[kk0 + kpat(j, kh)];
      cc[nt] = wmma_bf16(A, Bf, cc[nt]);
    }
  }
#pragma unroll
  for (int nt = 0; nt < 4; ++nt)
#pragma unroll
    for (int r = 0; r < 8; ++r) {
      const int M   = r + (kh << 3);
      const int nq  = n0 + M;
      const int col = (h << 6) + (nt << 4) + lm;
      concat[((size_t)b * 512 + nq) * 1536 + col] = cc[nt][r];
    }
}

// ---------------------------------------------------------------------------
// Edge context: per (b,n): ectx[h,:] = sum_m P[b,h,n,m] * k_edge[b,n,m,:]
// A = P rows (16 heads, 12 valid) x 512 keys; B = k_edge_t (b,n) 512x64.
// ---------------------------------------------------------------------------
__global__ __launch_bounds__(32)
void ectx_kernel(const unsigned short* __restrict__ Pws,
                 const unsigned short* __restrict__ ket,
                 float* __restrict__ ectx) {
  const int lane = threadIdx.x & 31;
  const int kh   = lane >> 4;
  const int lm   = lane & 15;
  const int bid  = blockIdx.x;          // B*L
  const unsigned short* Pbase = Pws + (size_t)bid * 16 * 512;
  const unsigned short* Kbase = ket + (size_t)bid * 64 * 512;
  v8f cc[4];
  for (int i = 0; i < 4; ++i)
    for (int e = 0; e < 8; ++e) cc[i][e] = 0.0f;
  for (int kc = 0; kc < 16; ++kc) {
    const int kk0 = kc << 5;
    BF16x16 A;
    const unsigned short* arow = Pbase + (size_t)lm * 512;
#pragma unroll
    for (int j = 0; j < 8; ++j)
      A.u[j] = *(const unsigned int*)&arow[kk0 + kpat(j, kh)];
#pragma unroll
    for (int nt = 0; nt < 4; ++nt) {
      const int d = (nt << 4) + lm;
      const unsigned short* brow = Kbase + (size_t)d * 512;
      BF16x16 Bf;
#pragma unroll
      for (int j = 0; j < 8; ++j)
        Bf.u[j] = *(const unsigned int*)&brow[kk0 + kpat(j, kh)];
      cc[nt] = wmma_bf16(A, Bf, cc[nt]);
    }
  }
#pragma unroll
  for (int nt = 0; nt < 4; ++nt)
#pragma unroll
    for (int r = 0; r < 8; ++r) {
      const int M = r + (kh << 3);
      if (M < H_) {
        const int col = (M << 6) + (nt << 4) + lm;
        ectx[(size_t)bid * 768 + col] = cc[nt][r];
      }
    }
}

// ---------------------------------------------------------------------------
extern "C" void kernel_launch(void* const* d_in, const int* in_sizes, int n_in,
                              void* d_out, int out_size, void* d_ws, size_t ws_size,
                              hipStream_t stream) {
  (void)in_sizes; (void)n_in; (void)out_size; (void)ws_size;
  const float* Q   = (const float*)d_in[0];
  const float* K   = (const float*)d_in[1];
  const float* V   = (const float*)d_in[2];
  const unsigned char* mask = (const unsigned char*)d_in[3];
  const float* E   = (const float*)d_in[4];
  const float* Wq  = (const float*)d_in[5];
  const float* bq  = (const float*)d_in[6];
  const float* Wk  = (const float*)d_in[7];
  const float* bk  = (const float*)d_in[8];
  const float* Wv  = (const float*)d_in[9];
  const float* bv  = (const float*)d_in[10];
  const float* Wke = (const float*)d_in[11];
  const float* bke = (const float*)d_in[12];
  const float* We  = (const float*)d_in[13];
  const float* be  = (const float*)d_in[14];
  const float* Weo = (const float*)d_in[15];
  const float* beo = (const float*)d_in[16];
  const float* Wo  = (const float*)d_in[17];
  const float* bo  = (const float*)d_in[18];

  char* ws = (char*)d_ws;
  size_t off = 0;
  auto alloc = [&](size_t bytes) -> void* {
    off = (off + 255) & ~(size_t)255;
    void* p = ws + off;
    off += bytes;
    return p;
  };
  unsigned short* q_ws  = (unsigned short*)alloc(2048ull * 768 * 2);
  unsigned short* k_ws  = (unsigned short*)alloc(2048ull * 768 * 2);
  unsigned short* v_ws  = (unsigned short*)alloc(2048ull * 768 * 2);
  unsigned short* ket   = (unsigned short*)alloc(4ull * 512 * 64 * 512 * 2);
  float*          eb    = (float*)alloc(4ull * 512 * 512 * 4);
  unsigned short* Pws   = (unsigned short*)alloc(4ull * 512 * 16 * 512 * 2);
  float*          ectx  = (float*)alloc(2048ull * 768 * 4);
  float*          cat   = (float*)alloc(2048ull * 1536 * 4);

  dim3 gg(6, 32), gb(256);
  gemm_bias_kernel<<<gg, gb, 0, stream>>>(Q, 768, Wq, 768, 768, bq, 0, q_ws, 0, 0);
  gemm_bias_kernel<<<gg, gb, 0, stream>>>(K, 768, Wk, 768, 768, bk, 0, k_ws, 0, 0);
  gemm_bias_kernel<<<gg, gb, 0, stream>>>(V, 768, Wv, 768, 768, bv, 1, v_ws, 0, 0);
  edge_kernel<<<16384, 256, 0, stream>>>(E, Wke, bke, We, be, ket, eb);
  attn_kernel<<<1536, 32, 0, stream>>>(q_ws, k_ws, v_ws, eb, mask, Pws, cat);
  ectx_kernel<<<2048, 32, 0, stream>>>(Pws, ket, ectx);
  gemm_bias_kernel<<<gg, gb, 0, stream>>>(ectx, 768, Weo, 768, 768, beo, 2, cat, 1536, 768);
  gemm_bias_kernel<<<gg, gb, 0, stream>>>(cat, 1536, Wo, 768, 1536, bo, 2, d_out, 768, 0);
}
